// LigerMoEExperts_42511586295841
// MI455X (gfx1250) — matrix operand order repaired and checked by reading, not compiled, not run on hardware
//
#include <hip/hip_runtime.h>
#include <hip/hip_bf16.h>
#include <cstdint>
#include <cstddef>

// Problem dims (fixed by the reference)
#define T_TOK 8192
#define D_DIM 1024
#define F_DIM 512
#define E_NUM 8
#define K_TOP 2

typedef __attribute__((ext_vector_type(16))) _Float16 v16h;
typedef __attribute__((ext_vector_type(8)))  _Float16 v8h;
typedef __attribute__((ext_vector_type(8)))  float    v8f;

union Frag16 { v16h v; v8h h[2]; };

// LDS tile geometry
#define BM 128         // assignment rows per block
#define BN 128         // output columns per block (gemm1: interleaved gate/up)
#define BK 32          // K per WMMA step
#define LDS_STRIDE 40  // halves per LDS row (pad 32 -> 40 to dodge bank conflicts)

// ---------------------------------------------------------------------------
// Zero output + routing scratch counters
// ---------------------------------------------------------------------------
__global__ void zero_init_kernel(float* __restrict__ out,
                                 int* __restrict__ counts,
                                 int* __restrict__ cursor) {
    size_t i = (size_t)blockIdx.x * blockDim.x + threadIdx.x;
    const size_t n4 = (size_t)T_TOK * D_DIM / 4;
    if (i < n4) {
        float4 z; z.x = 0.f; z.y = 0.f; z.z = 0.f; z.w = 0.f;
        ((float4*)out)[i] = z;
    }
    if (blockIdx.x == 0 && threadIdx.x < E_NUM) {
        counts[threadIdx.x] = 0;
        cursor[threadIdx.x] = 0;
    }
}

// ---------------------------------------------------------------------------
// Routing: count -> prefix -> fill compacted per-expert (token, weight) lists
// ---------------------------------------------------------------------------
__global__ void route_count_kernel(const int* __restrict__ tki,
                                   int* __restrict__ counts) {
    int t = blockIdx.x * blockDim.x + threadIdx.x;
    if (t >= T_TOK) return;
    int e0 = tki[t * K_TOP + 0];
    int e1 = tki[t * K_TOP + 1];
    atomicAdd(&counts[e0], 1);
    if (e1 != e0) atomicAdd(&counts[e1], 1);
}

__global__ void route_scan_kernel(const int* __restrict__ counts,
                                  int* __restrict__ bases) {
    if (blockIdx.x == 0 && threadIdx.x == 0) {
        int s = 0;
        for (int e = 0; e < E_NUM; ++e) { bases[e] = s; s += counts[e]; }
    }
}

__global__ void route_fill_kernel(const int* __restrict__ tki,
                                  const float* __restrict__ tkw,
                                  const int* __restrict__ bases,
                                  int* __restrict__ cursor,
                                  int* __restrict__ tok,
                                  float* __restrict__ wgt) {
    int t = blockIdx.x * blockDim.x + threadIdx.x;
    if (t >= T_TOK) return;
    int   e0 = tki[t * K_TOP + 0];
    int   e1 = tki[t * K_TOP + 1];
    float w0 = tkw[t * K_TOP + 0];
    float w1 = tkw[t * K_TOP + 1];
    if (e0 == e1) {
        int p = bases[e0] + atomicAdd(&cursor[e0], 1);
        tok[p] = t; wgt[p] = w0 + w1;
    } else {
        int p0 = bases[e0] + atomicAdd(&cursor[e0], 1);
        tok[p0] = t; wgt[p0] = w0;
        int p1 = bases[e1] + atomicAdd(&cursor[e1], 1);
        tok[p1] = t; wgt[p1] = w1;
    }
}

// ---------------------------------------------------------------------------
// GEMM1: act[row, f] = silu(x @ Wgate^T) * (x @ Wup^T), gathered rows, fp16 out
// Tile columns interleave gate/up: col 2j -> gate f0+j, col 2j+1 -> up f0+j,
// so SiLU pairing is a shfl_xor(1) within the WMMA accumulator layout.
// Grid: (64, F/64, E). Block: 256 threads = 8 wave32.
// ---------------------------------------------------------------------------
__global__ __launch_bounds__(256)
void moe_gemm1_silu_kernel(const float* __restrict__ hs,
                           const float* __restrict__ gup,
                           const int* __restrict__ counts,
                           const int* __restrict__ bases,
                           const int* __restrict__ tok,
                           _Float16* __restrict__ actws) {
    const int e   = blockIdx.z;
    const int cnt = counts[e];
    const int m0  = blockIdx.x * BM;
    if (m0 >= cnt) return;
    const int base = bases[e];
    const int f0   = blockIdx.y * (BN / 2);

    __shared__ _Float16 As[2][BM][LDS_STRIDE];
    __shared__ _Float16 Bs[2][BN][LDS_STRIDE];
    __shared__ int rowTok[BM];

    const int tid = threadIdx.x;

    if (tid < BM) {
        int gm = m0 + tid;
        rowTok[tid] = (gm < cnt) ? tok[base + gm] : 0;
    }
    __syncthreads();

    const float* __restrict__ wb = gup + (size_t)e * (2 * F_DIM) * D_DIM;

    auto loadTiles = [&](int buf, int k0) {
        // A: 128 rows x 32 cols fp32 -> fp16 (gathered by token id)
        #pragma unroll
        for (int j = 0; j < 4; ++j) {
            int idx = tid + 256 * j;          // 1024 tasks of float4
            int r   = idx >> 3;
            int cg  = idx & 7;
            const float* src = hs + (size_t)rowTok[r] * D_DIM + k0 + cg * 4;
            float4 x = *(const float4*)src;
            _Float16* dst = &As[buf][r][cg * 4];
            dst[0] = (_Float16)x.x; dst[1] = (_Float16)x.y;
            dst[2] = (_Float16)x.z; dst[3] = (_Float16)x.w;
        }
        // B: 128 interleaved gate/up weight rows x 32 cols fp32 -> fp16
        #pragma unroll
        for (int j = 0; j < 4; ++j) {
            int idx = tid + 256 * j;
            int n   = idx >> 3;
            int cg  = idx & 7;
            int f     = f0 + (n >> 1);
            int gurow = (n & 1) ? (F_DIM + f) : f;
            const float* src = wb + (size_t)gurow * D_DIM + k0 + cg * 4;
            if (k0 + BK < D_DIM) __builtin_prefetch(src + BK, 0, 0);
            float4 x = *(const float4*)src;
            _Float16* dst = &Bs[buf][n][cg * 4];
            dst[0] = (_Float16)x.x; dst[1] = (_Float16)x.y;
            dst[2] = (_Float16)x.z; dst[3] = (_Float16)x.w;
        }
    };

    const int lane = tid & 31;
    const int wid  = tid >> 5;
    const int wm   = wid & 3;   // 4 wave rows  -> 32 M each
    const int wn   = wid >> 2;  // 2 wave cols  -> 64 N each
    const int half = lane >> 4; // hi/lo 16 lanes
    const int l16  = lane & 15;

    v8f acc[2][4] = {};

    const int NSTEP = D_DIM / BK;  // 32
    loadTiles(0, 0);
    __syncthreads();

    for (int s = 0; s < NSTEP; ++s) {
        int buf = s & 1;
        if (s + 1 < NSTEP) loadTiles(buf ^ 1, (s + 1) * BK);

        Frag16 a[2], b[4];
        #pragma unroll
        for (int tm = 0; tm < 2; ++tm) {
            int row = wm * 32 + tm * 16 + l16;
            a[tm].h[0] = *(const v8h*)&As[buf][row][half * 16];
            a[tm].h[1] = *(const v8h*)&As[buf][row][half * 16 + 8];
        }
        #pragma unroll
        for (int tn = 0; tn < 4; ++tn) {
            int row = wn * 64 + tn * 16 + l16;
            b[tn].h[0] = *(const v8h*)&Bs[buf][row][half * 16];
            b[tn].h[1] = *(const v8h*)&Bs[buf][row][half * 16 + 8];
        }
        #pragma unroll
        for (int tm = 0; tm < 2; ++tm)
            #pragma unroll
            for (int tn = 0; tn < 4; ++tn)
                acc[tm][tn] = __builtin_amdgcn_wmma_f32_16x16x32_f16(
                    false, a[tm].v, false, b[tn].v,
                    (short)0, acc[tm][tn], false, false);
        __syncthreads();
    }

    // Epilogue: SiLU(gate) * up on even columns, store fp16 activations
    const int isGate = !(lane & 1);
    #pragma unroll
    for (int tm = 0; tm < 2; ++tm) {
        #pragma unroll
        for (int tn = 0; tn < 4; ++tn) {
            v8f c = acc[tm][tn];
            #pragma unroll
            for (int i = 0; i < 8; ++i) {
                float self  = c[i];
                float other = __shfl_xor(self, 1, 32);  // partner column
                if (isGate) {
                    float g = self, u = other;
                    float act = (g / (1.0f + __expf(-g))) * u;
                    int gm = m0 + wm * 32 + tm * 16 + half * 8 + i;
                    if (gm < cnt) {
                        int f = f0 + ((wn * 64 + tn * 16 + l16) >> 1);
                        actws[(size_t)(base + gm) * F_DIM + f] = (_Float16)act;
                    }
                }
            }
        }
    }
}

// ---------------------------------------------------------------------------
// GEMM2: out[token, d] += w_row * (act @ down[e]^T), fp16 A from ws,
// fp32->fp16 B on the fly, fp32 accum, float atomicAdd scatter (<=2 adds/elem
// onto zeroed base => bitwise deterministic).
// Grid: (64, D/128, E). Block: 256 threads = 8 wave32.
// ---------------------------------------------------------------------------
__global__ __launch_bounds__(256)
void moe_gemm2_down_kernel(const _Float16* __restrict__ actws,
                           const float* __restrict__ dwn,
                           const int* __restrict__ counts,
                           const int* __restrict__ bases,
                           const int* __restrict__ tok,
                           const float* __restrict__ wgt,
                           float* __restrict__ out) {
    const int e   = blockIdx.z;
    const int cnt = counts[e];
    const int m0  = blockIdx.x * BM;
    if (m0 >= cnt) return;
    const int base = bases[e];
    const int d0   = blockIdx.y * BN;

    __shared__ _Float16 As[2][BM][LDS_STRIDE];
    __shared__ _Float16 Bs[2][BN][LDS_STRIDE];
    __shared__ int   rowTok[BM];
    __shared__ float rowW[BM];

    const int tid = threadIdx.x;

    if (tid < BM) {
        int gm = m0 + tid;
        rowTok[tid] = (gm < cnt) ? tok[base + gm] : 0;
        rowW[tid]   = (gm < cnt) ? wgt[base + gm] : 0.0f;
    }
    __syncthreads();

    const float* __restrict__ wb = dwn + (size_t)e * D_DIM * F_DIM;

    auto loadTiles = [&](int buf, int k0) {
        // A: fp16 activation rows, straight 16B copies (512 tasks of 8 halves)
        #pragma unroll
        for (int j = 0; j < 2; ++j) {
            int idx = tid + 256 * j;
            int r   = idx >> 2;
            int g   = idx & 3;
            if (m0 + r < cnt) {
                const uint4 x = *(const uint4*)(actws +
                    (size_t)(base + m0 + r) * F_DIM + k0 + g * 8);
                *(uint4*)&As[buf][r][g * 8] = x;
            } else {
                uint4 z; z.x = 0u; z.y = 0u; z.z = 0u; z.w = 0u;
                *(uint4*)&As[buf][r][g * 8] = z;
            }
        }
        // B: down[e] rows (d columns of out), fp32 -> fp16
        #pragma unroll
        for (int j = 0; j < 4; ++j) {
            int idx = tid + 256 * j;
            int n   = idx >> 3;
            int cg  = idx & 7;
            const float* src = wb + (size_t)(d0 + n) * F_DIM + k0 + cg * 4;
            if (k0 + BK < F_DIM) __builtin_prefetch(src + BK, 0, 0);
            float4 x = *(const float4*)src;
            _Float16* dst = &Bs[buf][n][cg * 4];
            dst[0] = (_Float16)x.x; dst[1] = (_Float16)x.y;
            dst[2] = (_Float16)x.z; dst[3] = (_Float16)x.w;
        }
    };

    const int lane = tid & 31;
    const int wid  = tid >> 5;
    const int wm   = wid & 3;
    const int wn   = wid >> 2;
    const int half = lane >> 4;
    const int l16  = lane & 15;

    v8f acc[2][4] = {};

    const int NSTEP = F_DIM / BK;  // 16
    loadTiles(0, 0);
    __syncthreads();

    for (int s = 0; s < NSTEP; ++s) {
        int buf = s & 1;
        if (s + 1 < NSTEP) loadTiles(buf ^ 1, (s + 1) * BK);

        Frag16 a[2], b[4];
        #pragma unroll
        for (int tm = 0; tm < 2; ++tm) {
            int row = wm * 32 + tm * 16 + l16;
            a[tm].h[0] = *(const v8h*)&As[buf][row][half * 16];
            a[tm].h[1] = *(const v8h*)&As[buf][row][half * 16 + 8];
        }
        #pragma unroll
        for (int tn = 0; tn < 4; ++tn) {
            int row = wn * 64 + tn * 16 + l16;
            b[tn].h[0] = *(const v8h*)&Bs[buf][row][half * 16];
            b[tn].h[1] = *(const v8h*)&Bs[buf][row][half * 16 + 8];
        }
        #pragma unroll
        for (int tm = 0; tm < 2; ++tm)
            #pragma unroll
            for (int tn = 0; tn < 4; ++tn)
                acc[tm][tn] = __builtin_amdgcn_wmma_f32_16x16x32_f16(
                    false, a[tm].v, false, b[tn].v,
                    (short)0, acc[tm][tn], false, false);
        __syncthreads();
    }

    // Epilogue: scale by routing weight, atomic scatter into out[token, d]
    #pragma unroll
    for (int tm = 0; tm < 2; ++tm) {
        #pragma unroll
        for (int tn = 0; tn < 4; ++tn) {
            v8f c = acc[tm][tn];
            int d = d0 + wn * 64 + tn * 16 + l16;
            #pragma unroll
            for (int i = 0; i < 8; ++i) {
                int r  = wm * 32 + tm * 16 + half * 8 + i;
                int gm = m0 + r;
                if (gm < cnt) {
                    atomicAdd(&out[(size_t)rowTok[r] * D_DIM + d],
                              c[i] * rowW[r]);
                }
            }
        }
    }
}

// ---------------------------------------------------------------------------
// Launch
// ---------------------------------------------------------------------------
extern "C" void kernel_launch(void* const* d_in, const int* in_sizes, int n_in,
                              void* d_out, int out_size, void* d_ws, size_t ws_size,
                              hipStream_t stream) {
    (void)in_sizes; (void)n_in; (void)out_size; (void)ws_size;

    const float* hs  = (const float*)d_in[0];   // [T, D]
    const int*   tki = (const int*)d_in[1];     // [T, K]
    const float* tkw = (const float*)d_in[2];   // [T, K]
    const float* gup = (const float*)d_in[3];   // [E, 2F, D]
    const float* dwn = (const float*)d_in[4];   // [E, D, F]
    float*       out = (float*)d_out;           // [T, D]

    char* ws = (char*)d_ws;
    int*      counts = (int*)(ws + 0);
    int*      cursor = (int*)(ws + 256);
    int*      bases  = (int*)(ws + 512);
    int*      tokArr = (int*)(ws + 1024);
    float*    wgtArr = (float*)(ws + 1024 + sizeof(int) * T_TOK * K_TOP);
    _Float16* actws  = (_Float16*)(ws + (size_t)(1 << 18));  // 256 KB offset, 16 MB arena

    const size_t n4 = (size_t)T_TOK * D_DIM / 4;
    zero_init_kernel<<<dim3((unsigned)((n4 + 255) / 256)), 256, 0, stream>>>(out, counts, cursor);
    route_count_kernel<<<dim3(T_TOK / 256), 256, 0, stream>>>(tki, counts);
    route_scan_kernel<<<1, 32, 0, stream>>>(counts, bases);
    route_fill_kernel<<<dim3(T_TOK / 256), 256, 0, stream>>>(tki, tkw, bases, cursor, tokArr, wgtArr);

    moe_gemm1_silu_kernel<<<dim3(T_TOK / BM, F_DIM / (BN / 2), E_NUM), 256, 0, stream>>>(
        hs, gup, counts, bases, tokArr, actws);
    moe_gemm2_down_kernel<<<dim3(T_TOK / BM, D_DIM / BN, E_NUM), 256, 0, stream>>>(
        actws, dwn, counts, bases, tokArr, wgtArr, out);
}
